// EOQLinear_55173149884527
// MI455X (gfx1250) — matrix operand order repaired and checked by reading, not compiled, not run on hardware
//
#include <hip/hip_runtime.h>

typedef _Float16 v16h __attribute__((ext_vector_type(16)));
typedef _Float16 v8h  __attribute__((ext_vector_type(8)));
typedef float    v8f  __attribute__((ext_vector_type(8)));
typedef unsigned int v8u __attribute__((ext_vector_type(8)));
typedef unsigned int v4u __attribute__((ext_vector_type(4)));

#define K_PER_CHUNK 512   // K per workgroup pass (4 quant blocks of 128)
#define ROWS_PER_WG 128   // 8 waves x 16-row WMMA stripe
#define QBLK 128          // quant block size (matches reference)

// byte b (two nibbles) -> packed fp16 pair {1024+lo, 1024+hi}
// fp16 0x6400 = 1024.0; ulp in [1024,2048) is 1.0, so 1024+n is exact.
__device__ __forceinline__ unsigned pknib(unsigned b) {
  return (b & 0xFu) | ((b & 0xF0u) << 12) | 0x64006400u;
}

__global__ __launch_bounds__(256) void w4a16_gemv_wmma(
    const _Float16* __restrict__ x,       // [K] fp16
    const unsigned int* __restrict__ w,   // [N, K/2] one packed byte per int32
    const _Float16* __restrict__ scales,  // [N, K/QBLK] fp16
    float* __restrict__ ws,               // [N] fp32 partial accumulators
    int N, int K, int nblk)
{
  __shared__ __align__(32) _Float16 xs[K_PER_CHUNK];
  __shared__ __align__(16) _Float16 ss[ROWS_PER_WG * (K_PER_CHUNK / QBLK)];

  const int tid   = threadIdx.x;
  const int wave  = tid >> 5;
  const int lane  = tid & 31;
  const int laneM = lane & 15;   // row within 16-row stripe (A), column (B/D)
  const int hi    = lane >> 4;   // lane half selects K sub-range / D row half

  const int rowg0 = blockIdx.x * ROWS_PER_WG;               // first row of WG
  const int kc    = blockIdx.y * K_PER_CHUNK;               // K chunk base
  const int cb0   = blockIdx.y * (K_PER_CHUNK / QBLK);      // quant-block base

  // stage x chunk (512 halfs): 256 threads x 1 dword
  ((unsigned int*)xs)[tid] = ((const unsigned int*)(x + kc))[tid];
  // stage scales transposed: ss[bl*128 + r] = scales[rowg0+r, cb0+bl]
  #pragma unroll
  for (int i = 0; i < 2; ++i) {
    int idx = tid + i * 256;
    int bl  = idx >> 7;
    int r   = idx & 127;
    ss[idx] = scales[(size_t)(rowg0 + r) * nblk + cb0 + bl];
  }
  __syncthreads();

  const int row0 = rowg0 + wave * 16;
  const unsigned int* wrow = w + (size_t)(row0 + laneM) * (size_t)(K >> 1);

  v8f acc = {};  // scale-applied fp32 accumulator (16x16 C/D layout)

  #pragma unroll
  for (int bl = 0; bl < K_PER_CHUNK / QBLK; ++bl) {
    v8f cacc = {};  // raw integer-dot accumulator for this quant block
    #pragma unroll
    for (int t = 0; t < QBLK / 32; ++t) {
      const int kloc = bl * QBLK + t * 32;   // K offset within chunk
      const int c    = (kc + kloc) >> 1;     // int32 index within row

      // A tile (16x32 fp16 layout): lanes<16 take int32 [c..c+7],
      // lanes>=16 take [c+8..c+15]  -> two b128 streaming loads
      const v4u* ap = (const v4u*)(wrow + c + 8 * hi);
      v4u a0 = __builtin_nontemporal_load(ap);
      v4u a1 = __builtin_nontemporal_load(ap + 1);
      v8u au;
      au[0] = pknib(a0[0]); au[1] = pknib(a0[1]);
      au[2] = pknib(a0[2]); au[3] = pknib(a0[3]);
      au[4] = pknib(a1[0]); au[5] = pknib(a1[1]);
      au[6] = pknib(a1[2]); au[7] = pknib(a1[3]);
      v16h A = __builtin_bit_cast(v16h, au);
      A = A - (_Float16)1032.0f;             // (1024+n) - 1032 = n - 8

      // B tile: x replicated over 16 columns. lanes<16 hold K=0..15,
      // lanes>=16 hold K=16..31 -> one 32B LDS read (2x ds_load_b128)
      v8u bu = *(const v8u*)(&xs[kloc + 16 * hi]);
      v16h B = __builtin_bit_cast(v16h, bu);

      cacc = __builtin_amdgcn_wmma_f32_16x16x32_f16(
          false, A, false, B, (short)0, cacc, false, false);
    }
    // apply per-(row, block) scale: D VGPR v holds row (v + 8*hi)
    v8h sh = *(const v8h*)(&ss[bl * ROWS_PER_WG + wave * 16 + 8 * hi]);
    #pragma unroll
    for (int v = 0; v < 8; ++v)
      acc[v] += (float)sh[v] * cacc[v];
  }

  // All D columns are identical; lane 0 holds rows row0..row0+7,
  // lane 16 holds rows row0+8..row0+15.
  if (laneM == 0) {
    #pragma unroll
    for (int v = 0; v < 8; ++v)
      atomicAdd(&ws[row0 + hi * 8 + v], acc[v]);
  }
}

__global__ void zero_ws_kernel(float* __restrict__ ws, int n) {
  int i = blockIdx.x * blockDim.x + threadIdx.x;
  if (i < n) ws[i] = 0.0f;
}

__global__ void finalize_kernel(const float* __restrict__ ws,
                                _Float16* __restrict__ out, int n) {
  int i = blockIdx.x * blockDim.x + threadIdx.x;
  if (i < n) out[i] = (_Float16)ws[i];
}

extern "C" void kernel_launch(void* const* d_in, const int* in_sizes, int n_in,
                              void* d_out, int out_size, void* d_ws, size_t ws_size,
                              hipStream_t stream) {
  const _Float16*     x  = (const _Float16*)d_in[0];
  const unsigned int* w  = (const unsigned int*)d_in[1];
  const _Float16*     sc = (const _Float16*)d_in[2];

  const int       K     = in_sizes[0];
  const long long halfw = in_sizes[1];
  const int       N     = (int)((halfw * 2) / K);
  const int       nblk  = in_sizes[2] / N;   // K / QBLK

  float*    ws  = (float*)d_ws;
  _Float16* out = (_Float16*)d_out;

  zero_ws_kernel<<<(N + 255) / 256, 256, 0, stream>>>(ws, N);

  dim3 grid(N / ROWS_PER_WG, K / K_PER_CHUNK);
  w4a16_gemv_wmma<<<grid, 256, 0, stream>>>(x, w, sc, ws, N, K, nblk);

  finalize_kernel<<<(N + 255) / 256, 256, 0, stream>>>(ws, out, N);
}
